// LightweightMamba2MoE_33432025432262
// MI455X (gfx1250) — matrix operand order repaired and checked by reading, not compiled, not run on hardware
//
#include <hip/hip_runtime.h>

// ---------------- types ----------------
typedef __attribute__((ext_vector_type(8)))  __bf16 v8bf;
typedef __attribute__((ext_vector_type(16))) __bf16 v16bf;
typedef __attribute__((ext_vector_type(8)))  float  v8f;

union Frag { v16bf v; v8bf h[2]; __bf16 s[16]; };
union BU   { v8bf v[2]; __bf16 s[16]; };

#define T_TOK   8192      // B*L
#define D_MODEL 512
#define D_INNER 1024
#define D_STATE 128
#define NHEADS  16
#define HEADDIM 64
#define DPROJ   2320      // 2*D_INNER + 2*D_STATE + NHEADS
#define DPROJ_P 2432      // padded to 19*128 so the GEMM needs no N guards
#define CONVD   1280      // D_INNER + 2*D_STATE
#define E_HID   2048
#define NEXP    8

__device__ __forceinline__ float siluf(float v) { return v / (1.f + __expf(-v)); }

__device__ __forceinline__ float block_reduce_sum_256(float v, float* red8) {
  #pragma unroll
  for (int o = 16; o > 0; o >>= 1) v += __shfl_xor(v, o, 32);
  int w = threadIdx.x >> 5;
  if ((threadIdx.x & 31) == 0) red8[w] = v;
  __syncthreads();
  if (threadIdx.x == 0) {
    float s = 0.f;
    #pragma unroll
    for (int i = 0; i < 8; ++i) s += red8[i];
    red8[0] = s;
  }
  __syncthreads();
  float out = red8[0];
  __syncthreads();
  return out;
}

// ---------------- elementwise ----------------
__global__ void __launch_bounds__(256) embed_kernel(const int* __restrict__ tokens,
    const float* __restrict__ emb, const float* __restrict__ pos, float* __restrict__ x) {
  int idx = blockIdx.x * 256 + threadIdx.x;
  if (idx >= T_TOK * D_MODEL) return;
  int d = idx & (D_MODEL - 1);
  int t = idx >> 9;
  int l = t & 1023;
  x[idx] = emb[(size_t)tokens[t] * D_MODEL + d] + pos[l * D_MODEL + d];
}

__global__ void __launch_bounds__(256) cvt_bf16_kernel(const float* __restrict__ src,
                                                       __bf16* __restrict__ dst, int n) {
  int i = blockIdx.x * 256 + threadIdx.x;
  if (i < n) dst[i] = (__bf16)src[i];
}

// row-major convert with zero-padded extra columns (ncol_src -> ncol_dst)
__global__ void __launch_bounds__(256) cvt_pad_bf16_kernel(const float* __restrict__ src,
    __bf16* __restrict__ dst, int rows, int ncol_src, int ncol_dst) {
  int i = blockIdx.x * 256 + threadIdx.x;
  if (i >= rows * ncol_dst) return;
  int r = i / ncol_dst;
  int c = i - r * ncol_dst;
  dst[i] = (c < ncol_src) ? (__bf16)src[(size_t)r * ncol_src + c] : (__bf16)0.f;
}

__global__ void __launch_bounds__(256) zero_kernel(float* p, int n) {
  int i = blockIdx.x * 256 + threadIdx.x;
  if (i < n) p[i] = 0.f;
}

// ---------------- WMMA bf16 GEMM ----------------
// C[M,N] = A[M,K] @ B[K,N], fp32 accum.
// Requires M%128==0, N%128==0, K%32==0 (all call sites satisfy this).
// 256 threads = 8 waves; block tile 128x128; wave tile 32x64 (2x4 wmma);
// LDS double-buffered, one barrier per K-step; B staged transposed so both
// A and B fragments are pairs of aligned ds_load_b128.
__global__ void __launch_bounds__(256) gemm_bf16_wmma_kernel(
    const __bf16* __restrict__ A, const __bf16* __restrict__ B,
    float* __restrict__ C, int M, int N, int K)
{
  __shared__ __align__(16) __bf16 As[2][128][32];   // [m][k]         16 KB
  __shared__ __align__(16) __bf16 BsT[2][128][40];  // [n][k] (+pad)  20 KB

  const int tid  = threadIdx.x;
  const int lane = tid & 31;
  const int wave = tid >> 5;
  const int wm   = (wave >> 1) * 32;   // 0,32,64,96
  const int wn   = (wave & 1) * 64;    // 0,64
  const int m0   = blockIdx.y * 128;
  const int n0   = blockIdx.x * 128;

  // staging assignments
  const int ar = tid >> 1;             // 0..127 (A row)
  const int ac = (tid & 1) << 4;       // 0/16   (A col base, 16 els)
  const int br = tid >> 3;             // 0..31  (B row = k)
  const int bc = (tid & 7) << 4;       // 0..112 (B col base, 16 els)

  const __bf16* gA = A + (size_t)(m0 + ar) * K + ac;
  const __bf16* gB = B + (size_t)br * N + n0 + bc;

  BU pa, pb;
  pa.v[0] = *(const v8bf*)(gA);
  pa.v[1] = *(const v8bf*)(gA + 8);
  pb.v[0] = *(const v8bf*)(gB);
  pb.v[1] = *(const v8bf*)(gB + 8);
  *(v8bf*)&As[0][ar][ac]     = pa.v[0];
  *(v8bf*)&As[0][ar][ac + 8] = pa.v[1];
  #pragma unroll
  for (int j = 0; j < 16; ++j) BsT[0][bc + j][br] = pb.s[j];
  __syncthreads();

  v8f acc[2][4];
  #pragma unroll
  for (int mi = 0; mi < 2; ++mi)
    #pragma unroll
    for (int ni = 0; ni < 4; ++ni)
      #pragma unroll
      for (int e = 0; e < 8; ++e) acc[mi][ni][e] = 0.f;

  // fragment lane mapping (CDNA5 wave32 WMMA layouts)
  const int rowA = lane & 15;
  const int kq   = (lane >> 4) << 3;   // A: hi half-wave holds K+8 block
  const int kb   = (lane >> 4) << 3;   // B: hi half-wave holds K+8 block
  const int col  = lane & 15;

  const int nIter = K >> 5;
  int cur = 0;
  for (int it = 0; it < nIter; ++it) {
    const bool more = (it + 1 < nIter);
    if (more) {
      const __bf16* ga = gA + (size_t)(it + 1) * 32;
      const __bf16* gb = gB + (size_t)(it + 1) * 32 * N;
      pa.v[0] = *(const v8bf*)(ga);
      pa.v[1] = *(const v8bf*)(ga + 8);
      pb.v[0] = *(const v8bf*)(gb);
      pb.v[1] = *(const v8bf*)(gb + 8);
    }

    Frag a[2], b[4];
    #pragma unroll
    for (int mi = 0; mi < 2; ++mi) {
      const __bf16* p = &As[cur][wm + mi * 16 + rowA][0];
      a[mi].h[0] = *(const v8bf*)(p + kq);
      a[mi].h[1] = *(const v8bf*)(p + 16 + kq);
    }
    #pragma unroll
    for (int ni = 0; ni < 4; ++ni) {
      const __bf16* p = &BsT[cur][wn + ni * 16 + col][0];
      b[ni].h[0] = *(const v8bf*)(p + kb);
      b[ni].h[1] = *(const v8bf*)(p + 16 + kb);
    }
    #pragma unroll
    for (int mi = 0; mi < 2; ++mi)
      #pragma unroll
      for (int ni = 0; ni < 4; ++ni)
        acc[mi][ni] = __builtin_amdgcn_wmma_f32_16x16x32_bf16(
            false, a[mi].v, false, b[ni].v, (short)0, acc[mi][ni], false, false);

    if (more) {
      const int nxt = cur ^ 1;
      *(v8bf*)&As[nxt][ar][ac]     = pa.v[0];
      *(v8bf*)&As[nxt][ar][ac + 8] = pa.v[1];
      #pragma unroll
      for (int j = 0; j < 16; ++j) BsT[nxt][bc + j][br] = pb.s[j];
    }
    __syncthreads();
    cur ^= 1;
  }

  // store: C 16x16 f32 layout (VGPR r -> M=r lanes 0-15, M=8+r lanes 16-31)
  const int rbase = (lane >> 4) << 3;
  const int cl    = lane & 15;
  #pragma unroll
  for (int mi = 0; mi < 2; ++mi)
    #pragma unroll
    for (int ni = 0; ni < 4; ++ni) {
      const int crow = m0 + wm + mi * 16 + rbase;
      const int ccol = n0 + wn + ni * 16 + cl;
      #pragma unroll
      for (int r = 0; r < 8; ++r)
        C[(size_t)(crow + r) * N + ccol] = acc[mi][ni][r];
    }
}

// ---------------- mamba pieces ----------------
__global__ void __launch_bounds__(256) dt_kernel(const float* __restrict__ zx,
    const float* __restrict__ dt_bias, const float* __restrict__ A_log,
    float* __restrict__ dt, float* __restrict__ dA) {
  int i = blockIdx.x * 256 + threadIdx.x;
  if (i >= T_TOK * NHEADS) return;
  int h = i & 15;
  int t = i >> 4;
  float v = zx[(size_t)t * DPROJ_P + 2304 + h] + dt_bias[h];
  float sp = (v > 20.f) ? v : log1pf(__expf(v));
  float A = -__expf(A_log[h]);
  dt[i] = sp;
  dA[i] = __expf(sp * A);
}

__global__ void __launch_bounds__(256) conv_silu_kernel(const float* __restrict__ zx,
    const float* __restrict__ conv_w, const float* __restrict__ conv_b,
    float* __restrict__ xBC) {
  int idx = blockIdx.x * 256 + threadIdx.x;
  if (idx >= T_TOK * CONVD) return;
  int c = idx % CONVD;
  int t = idx / CONVD;
  int l = t & 1023;
  float sum = 0.f;
  #pragma unroll
  for (int k = 0; k < 4; ++k) {
    int ls = l - 3 + k;
    if (ls >= 0)
      sum += zx[(size_t)(t - l + ls) * DPROJ_P + D_INNER + c] * conv_w[c * 4 + k];
  }
  float v = sum + conv_b[c];
  xBC[(size_t)t * CONVD + c] = siluf(v);
}

// one block per (batch, head); state [64][128] in registers (32 floats/thread)
__global__ void __launch_bounds__(256) ssm_scan_kernel(const float* __restrict__ xBC,
    const float* __restrict__ dt, const float* __restrict__ dA,
    const float* __restrict__ Dvec, float* __restrict__ y) {
  int bh = blockIdx.x;
  int b = bh >> 4;
  int h = bh & 15;
  int tid = threadIdx.x;
  int p  = tid >> 2;              // headdim row 0..63
  int ng = (tid & 3) * 32;        // state chunk base
  float s[32];
  #pragma unroll
  for (int j = 0; j < 32; ++j) s[j] = 0.f;
  __shared__ float Bs[128], Cs[128], xs[64];
  const float Dh = Dvec[h];
  for (int l = 0; l < 1024; ++l) {
    const float* row = xBC + ((size_t)b * 1024 + l) * CONVD;
    if (tid < 128) Bs[tid] = row[D_INNER + tid];
    else           Cs[tid - 128] = row[D_INNER + D_STATE + (tid - 128)];
    if (tid < 64)  xs[tid] = row[h * 64 + tid];
    __syncthreads();
    float dt_t = dt[((size_t)b * 1024 + l) * NHEADS + h];
    float dA_t = dA[((size_t)b * 1024 + l) * NHEADS + h];
    float dtx = dt_t * xs[p];
    float acc = 0.f;
    #pragma unroll
    for (int j = 0; j < 32; ++j) {
      s[j] = s[j] * dA_t + dtx * Bs[ng + j];
      acc += s[j] * Cs[ng + j];
    }
    acc += __shfl_xor(acc, 1, 32);
    acc += __shfl_xor(acc, 2, 32);
    if ((tid & 3) == 0)
      y[(((size_t)b * 1024 + l) * NHEADS + h) * 64 + p] = acc + Dh * xs[p];
    __syncthreads();
  }
}

__global__ void __launch_bounds__(256) gated_rms_kernel(const float* __restrict__ y,
    const float* __restrict__ zx, const float* __restrict__ nw, __bf16* __restrict__ out) {
  int tok = blockIdx.x;
  __shared__ float red[8];
  float v[4]; float ss = 0.f;
  #pragma unroll
  for (int i = 0; i < 4; ++i) {
    int d = threadIdx.x + i * 256;
    float z = zx[(size_t)tok * DPROJ_P + d];
    float val = y[(size_t)tok * D_INNER + d] * siluf(z);
    v[i] = val; ss += val * val;
  }
  float tot = block_reduce_sum_256(ss, red);
  float scale = rsqrtf(tot * (1.f / D_INNER) + 1e-5f);
  #pragma unroll
  for (int i = 0; i < 4; ++i) {
    int d = threadIdx.x + i * 256;
    out[(size_t)tok * D_INNER + d] = (__bf16)(v[i] * scale * nw[d]);
  }
}

// x += LN(t; s, b); also refresh bf16 copy of x
__global__ void __launch_bounds__(256) residual_ln_kernel(float* __restrict__ x,
    __bf16* __restrict__ xbf, const float* __restrict__ t,
    const float* __restrict__ s, const float* __restrict__ b) {
  int tok = blockIdx.x;
  __shared__ float red[8];
  const float* r = t + (size_t)tok * D_MODEL;
  int d0 = threadIdx.x, d1 = threadIdx.x + 256;
  float v0 = r[d0], v1 = r[d1];
  float mean = block_reduce_sum_256(v0 + v1, red) * (1.f / D_MODEL);
  float dv0 = v0 - mean, dv1 = v1 - mean;
  float var = block_reduce_sum_256(dv0 * dv0 + dv1 * dv1, red) * (1.f / D_MODEL);
  float rs = rsqrtf(var + 1e-5f);
  float o0 = x[(size_t)tok * D_MODEL + d0] + dv0 * rs * s[d0] + b[d0];
  float o1 = x[(size_t)tok * D_MODEL + d1] + dv1 * rs * s[d1] + b[d1];
  x[(size_t)tok * D_MODEL + d0] = o0;
  x[(size_t)tok * D_MODEL + d1] = o1;
  xbf[(size_t)tok * D_MODEL + d0] = (__bf16)o0;
  xbf[(size_t)tok * D_MODEL + d1] = (__bf16)o1;
}

// ---------------- MoE ----------------
__global__ void __launch_bounds__(256) gate_topk_kernel(const float* __restrict__ x,
    const float* __restrict__ gs, const float* __restrict__ gb,
    const float* __restrict__ gw, const float* __restrict__ gbias,
    float* __restrict__ coef) {
  int tok = blockIdx.x;
  __shared__ float red[8];
  __shared__ float ln[D_MODEL];
  __shared__ float lg[NEXP];
  const float* r = x + (size_t)tok * D_MODEL;
  int d0 = threadIdx.x, d1 = threadIdx.x + 256;
  float v0 = r[d0], v1 = r[d1];
  float mean = block_reduce_sum_256(v0 + v1, red) * (1.f / D_MODEL);
  float dv0 = v0 - mean, dv1 = v1 - mean;
  float var = block_reduce_sum_256(dv0 * dv0 + dv1 * dv1, red) * (1.f / D_MODEL);
  float rs = rsqrtf(var + 1e-5f);
  ln[d0] = dv0 * rs * gs[d0] + gb[d0];
  ln[d1] = dv1 * rs * gs[d1] + gb[d1];
  __syncthreads();
  if (threadIdx.x < NEXP) {
    float acc = gbias[threadIdx.x];
    for (int d = 0; d < D_MODEL; ++d) acc += ln[d] * gw[d * NEXP + threadIdx.x];
    lg[threadIdx.x] = acc;
  }
  __syncthreads();
  if (threadIdx.x == 0) {
    float mx = lg[0];
    #pragma unroll
    for (int e = 1; e < NEXP; ++e) mx = fmaxf(mx, lg[e]);
    float p[NEXP], ps = 0.f;
    #pragma unroll
    for (int e = 0; e < NEXP; ++e) { p[e] = __expf(lg[e] - mx); ps += p[e]; }
    #pragma unroll
    for (int e = 0; e < NEXP; ++e) p[e] /= ps;
    int i0 = 0;
    #pragma unroll
    for (int e = 1; e < NEXP; ++e) if (p[e] > p[i0]) i0 = e;
    int i1 = (i0 == 0) ? 1 : 0;
    #pragma unroll
    for (int e = 0; e < NEXP; ++e) if (e != i0 && p[e] > p[i1]) i1 = e;
    float s2 = p[i0] + p[i1];
    #pragma unroll
    for (int e = 0; e < NEXP; ++e) coef[tok * NEXP + e] = 0.f;
    coef[tok * NEXP + i0] = p[i0] / s2;
    coef[tok * NEXP + i1] = p[i1] / s2;
  }
}

__global__ void __launch_bounds__(256) gelu_cvt_kernel(const float* __restrict__ hin,
    const float* __restrict__ b1, __bf16* __restrict__ hout, int n, int H) {
  int i = blockIdx.x * 256 + threadIdx.x;
  if (i >= n) return;
  int c = i % H;
  float v = hin[i] + b1[c];
  float g = 0.5f * v * (1.f + erff(v * 0.70710678118654752f));
  hout[i] = (__bf16)g;
}

__global__ void __launch_bounds__(256) moe_accum_kernel(const float* __restrict__ eo,
    const float* __restrict__ b2, const float* __restrict__ coef, int e,
    float* __restrict__ out) {
  int i = blockIdx.x * 256 + threadIdx.x;
  if (i >= T_TOK * D_MODEL) return;
  int d = i & (D_MODEL - 1);
  int t = i >> 9;
  float c = coef[t * NEXP + e];
  out[i] += c * (eo[i] + b2[d]);
}

// ---------------- pooling + head ----------------
__global__ void __launch_bounds__(256) mask_kernel(const float* __restrict__ x,
                                                   float* __restrict__ flags) {
  int tok = blockIdx.x;
  __shared__ float red[8];
  const float* r = x + (size_t)tok * D_MODEL;
  float s = r[threadIdx.x] + r[threadIdx.x + 256];
  float tot = block_reduce_sum_256(s, red);
  if (threadIdx.x == 0) flags[tok] = (tot != 0.f) ? 1.f : 0.f;
}

__global__ void __launch_bounds__(512) pool_kernel(const float* __restrict__ x,
    const float* __restrict__ flags, float* __restrict__ pooled) {
  int b = blockIdx.x;
  int d = threadIdx.x;
  float s = 0.f, c = 0.f;
  for (int l = 0; l < 1024; ++l) {
    float f = flags[b * 1024 + l];
    s += x[((size_t)b * 1024 + l) * D_MODEL + d] * f;
    c += f;
  }
  pooled[b * D_MODEL + d] = s / fmaxf(c, 1.f);
}

__global__ void __launch_bounds__(128) head_kernel(const float* __restrict__ pooled,
    const float* __restrict__ w1, const float* __restrict__ b1,
    const float* __restrict__ w2, const float* __restrict__ b2,
    float* __restrict__ out) {
  int b = blockIdx.x;
  __shared__ float h[128];
  float acc = b1[threadIdx.x];
  for (int d = 0; d < D_MODEL; ++d) acc += pooled[b * D_MODEL + d] * w1[d * 128 + threadIdx.x];
  h[threadIdx.x] = fmaxf(acc, 0.f);
  __syncthreads();
  if (threadIdx.x < 2) {
    float o = b2[threadIdx.x];
    for (int j = 0; j < 128; ++j) o += h[j] * w2[j * 2 + threadIdx.x];
    out[b * 2 + threadIdx.x] = o;
  }
}

// ---------------- launcher ----------------
extern "C" void kernel_launch(void* const* d_in, const int* in_sizes, int n_in,
                              void* d_out, int out_size, void* d_ws, size_t ws_size,
                              hipStream_t stream) {
  (void)in_sizes; (void)n_in; (void)out_size; (void)ws_size;
  const int*   tokens = (const int*)  d_in[0];
  const float* emb    = (const float*)d_in[1];
  const float* pos    = (const float*)d_in[2];

  char* w = (char*)d_ws;
  auto carve = [&](size_t bytes) { char* p = w; w += (bytes + 255) & ~(size_t)255; return p; };
  float*  x      = (float*) carve((size_t)T_TOK * D_MODEL * 4);
  __bf16* xbf    = (__bf16*)carve((size_t)T_TOK * D_MODEL * 2);
  float*  zx     = (float*) carve((size_t)T_TOK * DPROJ_P * 4);
  float*  xBC    = (float*) carve((size_t)T_TOK * CONVD * 4);
  float*  dtb    = (float*) carve((size_t)T_TOK * NHEADS * 4);
  float*  dab    = (float*) carve((size_t)T_TOK * NHEADS * 4);
  float*  yssm   = (float*) carve((size_t)T_TOK * D_INNER * 4);
  __bf16* ygate  = (__bf16*)carve((size_t)T_TOK * D_INNER * 2);
  float*  tmp    = (float*) carve((size_t)T_TOK * D_MODEL * 4);
  float*  moeo   = (float*) carve((size_t)T_TOK * D_MODEL * 4);
  float*  hbuf   = (float*) carve((size_t)T_TOK * E_HID * 4);
  __bf16* hbf    = (__bf16*)carve((size_t)T_TOK * E_HID * 2);
  __bf16* wbuf   = (__bf16*)carve((size_t)D_MODEL * DPROJ_P * 2);
  float*  coef   = (float*) carve((size_t)T_TOK * NEXP * 4);
  float*  flags  = (float*) carve((size_t)T_TOK * 4);
  float*  pooled = (float*) carve((size_t)8 * D_MODEL * 4);

  dim3 b256(256);
  embed_kernel<<<dim3((T_TOK * D_MODEL) / 256), b256, 0, stream>>>(tokens, emb, pos, x);
  cvt_bf16_kernel<<<dim3((T_TOK * D_MODEL) / 256), b256, 0, stream>>>(x, xbf, T_TOK * D_MODEL);

  for (int layer = 0; layer < 2; ++layer) {
    int base = 3 + layer * 20;
    const float* in_proj_w = (const float*)d_in[base + 0];
    const float* conv_w    = (const float*)d_in[base + 1];
    const float* conv_b    = (const float*)d_in[base + 2];
    const float* dt_bias   = (const float*)d_in[base + 3];
    const float* A_log     = (const float*)d_in[base + 4];
    const float* Dvec      = (const float*)d_in[base + 5];
    const float* norm_w    = (const float*)d_in[base + 6];
    const float* out_projw = (const float*)d_in[base + 7];
    const float* gate_ln_s = (const float*)d_in[base + 8];
    const float* gate_ln_b = (const float*)d_in[base + 9];
    const float* gate_w    = (const float*)d_in[base + 10];
    const float* gate_b    = (const float*)d_in[base + 11];
    const float* w1        = (const float*)d_in[base + 12];
    const float* b1        = (const float*)d_in[base + 13];
    const float* w2        = (const float*)d_in[base + 14];
    const float* b2        = (const float*)d_in[base + 15];
    const float* nm_s      = (const float*)d_in[base + 16];
    const float* nm_b      = (const float*)d_in[base + 17];
    const float* nmoe_s    = (const float*)d_in[base + 18];
    const float* nmoe_b    = (const float*)d_in[base + 19];

    // ---- Mamba2 ----
    cvt_pad_bf16_kernel<<<dim3((D_MODEL * DPROJ_P) / 256), b256, 0, stream>>>(
        in_proj_w, wbuf, D_MODEL, DPROJ, DPROJ_P);
    gemm_bf16_wmma_kernel<<<dim3(DPROJ_P / 128, T_TOK / 128), b256, 0, stream>>>(
        xbf, wbuf, zx, T_TOK, DPROJ_P, D_MODEL);
    dt_kernel<<<dim3((T_TOK * NHEADS) / 256), b256, 0, stream>>>(zx, dt_bias, A_log, dtb, dab);
    conv_silu_kernel<<<dim3((T_TOK * CONVD) / 256), b256, 0, stream>>>(zx, conv_w, conv_b, xBC);
    ssm_scan_kernel<<<dim3(8 * NHEADS), b256, 0, stream>>>(xBC, dtb, dab, Dvec, yssm);
    gated_rms_kernel<<<dim3(T_TOK), b256, 0, stream>>>(yssm, zx, norm_w, ygate);
    cvt_bf16_kernel<<<dim3((D_INNER * D_MODEL) / 256), b256, 0, stream>>>(out_projw, wbuf, D_INNER * D_MODEL);
    gemm_bf16_wmma_kernel<<<dim3(D_MODEL / 128, T_TOK / 128), b256, 0, stream>>>(
        ygate, wbuf, tmp, T_TOK, D_MODEL, D_INNER);
    residual_ln_kernel<<<dim3(T_TOK), b256, 0, stream>>>(x, xbf, tmp, nm_s, nm_b);

    // ---- MoE (top-2 via zeroed coefficients) ----
    gate_topk_kernel<<<dim3(T_TOK), b256, 0, stream>>>(x, gate_ln_s, gate_ln_b, gate_w, gate_b, coef);
    zero_kernel<<<dim3((T_TOK * D_MODEL) / 256), b256, 0, stream>>>(moeo, T_TOK * D_MODEL);
    for (int e = 0; e < NEXP; ++e) {
      cvt_bf16_kernel<<<dim3((D_MODEL * E_HID) / 256), b256, 0, stream>>>(
          w1 + (size_t)e * D_MODEL * E_HID, wbuf, D_MODEL * E_HID);
      gemm_bf16_wmma_kernel<<<dim3(E_HID / 128, T_TOK / 128), b256, 0, stream>>>(
          xbf, wbuf, hbuf, T_TOK, E_HID, D_MODEL);
      gelu_cvt_kernel<<<dim3((T_TOK * E_HID) / 256), b256, 0, stream>>>(
          hbuf, b1 + e * E_HID, hbf, T_TOK * E_HID, E_HID);
      cvt_bf16_kernel<<<dim3((E_HID * D_MODEL) / 256), b256, 0, stream>>>(
          w2 + (size_t)e * E_HID * D_MODEL, wbuf, E_HID * D_MODEL);
      gemm_bf16_wmma_kernel<<<dim3(D_MODEL / 128, T_TOK / 128), b256, 0, stream>>>(
          hbf, wbuf, tmp, T_TOK, D_MODEL, E_HID);
      moe_accum_kernel<<<dim3((T_TOK * D_MODEL) / 256), b256, 0, stream>>>(
          tmp, b2 + e * D_MODEL, coef, e, moeo);
    }
    residual_ln_kernel<<<dim3(T_TOK), b256, 0, stream>>>(x, xbf, moeo, nmoe_s, nmoe_b);
  }

  mask_kernel<<<dim3(T_TOK), b256, 0, stream>>>(x, flags);
  pool_kernel<<<dim3(8), dim3(512), 0, stream>>>(x, flags, pooled);
  head_kernel<<<dim3(8), dim3(128), 0, stream>>>(pooled,
      (const float*)d_in[43], (const float*)d_in[44],
      (const float*)d_in[45], (const float*)d_in[46], (float*)d_out);
}
